// ConditionalRowRNN_82429012345536
// MI455X (gfx1250) — matrix operand order repaired and checked by reading, not compiled, not run on hardware
//
#include <hip/hip_runtime.h>

#define BD 16
#define CH 3
#define RD 64
#define SD 64
#define HD 32
#define ED 64
#define VD 258
#define NH 96   // 3*H

typedef __attribute__((ext_vector_type(16))) _Float16 v16h;
typedef __attribute__((ext_vector_type(8)))  _Float16 v8h;
typedef __attribute__((ext_vector_type(4)))  _Float16 v4h;
typedef __attribute__((ext_vector_type(8)))  float    v8f;
typedef __attribute__((ext_vector_type(4)))  float    f4v;

// ---------------- WMMA + fragment helpers (CDNA5 wave32 layouts) ----------------

static __device__ __forceinline__ v8f wmma16(v16h a, v16h b, v8f c) {
  // D = A(16x32 f16) * B(32x16 f16) + C(16x16 f32)
  return __builtin_amdgcn_wmma_f32_16x16x32_f16(false, a, false, b, (short)0, c, false, false);
}

// A-matrix 16x32 f16: lane L holds row m=L&15; K chunks [8*kb,8*kb+8) and [16+8*kb,+8)
static __device__ __forceinline__ v16h load_a16(const _Float16* p, int ldk, int k0, int lane) {
  const int m = lane & 15, kb = lane >> 4;
  const _Float16* q0 = p + m * ldk + k0 + 8 * kb;
  v8h lo = *(const v8h*)q0;          // 16B vector load -> b128
  v8h hi = *(const v8h*)(q0 + 16);
  v16h a;
#pragma unroll
  for (int i = 0; i < 8; ++i) { a[i] = lo[i]; a[i + 8] = hi[i]; }
  return a;
}

// B-matrix 32x16 f16: lane L holds column n=L&15; K chunks [8*kb,8*kb+8) and [16+8*kb,+8)
// With W stored row-major [n_out][k], B[k][n] = W[n][k] -> natural row read.
static __device__ __forceinline__ v16h load_b16(const _Float16* p, int ldk, int nt, int k0, int lane) {
  const int n = lane & 15, kb = lane >> 4;
  const _Float16* q = p + (nt * 16 + n) * ldk + k0 + 8 * kb;
  v8h lo = *(const v8h*)q;
  v8h hi = *(const v8h*)(q + 16);
  v16h b;
#pragma unroll
  for (int i = 0; i < 8; ++i) { b[i] = lo[i]; b[i + 8] = hi[i]; }
  return b;
}

// C/D 16x16 f32 tile -> f16 row-major [16][ldk] store (lane = n, VGPR v = m within half)
static __device__ __forceinline__ void store_c16(_Float16* p, int ldk, int nt, v8f c, int lane) {
  const int n = lane & 15, mh = (lane >> 4) * 8;
#pragma unroll
  for (int v = 0; v < 8; ++v) p[(mh + v) * ldk + nt * 16 + n] = (_Float16)c[v];
}

static __device__ __forceinline__ v8f splat8(float x) {
  v8f r;
#pragma unroll
  for (int i = 0; i < 8; ++i) r[i] = x;
  return r;
}

static __device__ __forceinline__ v8f sigm8(v8f x) {
  v8f r;
#pragma unroll
  for (int i = 0; i < 8; ++i) r[i] = 1.f / (1.f + __expf(-x[i]));
  return r;
}

static __device__ __forceinline__ float tanh_fast(float x) {
  return 2.f / (1.f + __expf(-2.f * x)) - 1.f;
}

// ---------------- init: zero flags + zero-ph block ----------------

__global__ void init_kernel(unsigned* flags, _Float16* zeroph) {
  int i = blockIdx.x * blockDim.x + threadIdx.x;
  if (i < CH * RD * SD) flags[i] = 0u;
  if (i < 512) zeroph[i] = (_Float16)0.f;
}

// ---------------- embedding gather (x and target) ----------------
// exbuf cell layout is C-tile order: [tile t=0..4][lane=n+16*(m>>3)][v=m&7]
// so the RNN's comb C-init is one 16B load per tile per lane.

__global__ __launch_bounds__(256, 1) void embed_kernel(
    const int* __restrict__ x, const int* __restrict__ target,
    const float* __restrict__ emb,
    _Float16* __restrict__ exbuf,   // [3][RD][SD][5][32][8]
    _Float16* __restrict__ tgtbuf)  // [2][RD][SD][16][64]  red_e / green_e
{
  int id = blockIdx.x * blockDim.x + threadIdx.x;
  if (id >= CH * RD * SD * BD) return;
  const int b = id & 15; int t = id >> 4;
  const int s = t & 63; t >>= 6;
  const int r = t & 63; t >>= 6;
  const int c = t;

  const int xv = x[((b * CH + c) * RD + r) * SD + s];
  const float* e = emb + ((size_t)c * VD + xv) * ED;

  _Float16* dst = exbuf + (size_t)((c * RD + r) * SD + s) * 1280;
  const int laneHalf = (b >> 3) * 16;   // 0 or 16
  const int vv = b & 7;
#pragma unroll 4
  for (int col = 0; col < ED; ++col) {
    const int tt = col >> 4, nn = col & 15;
    dst[((tt * 32) + nn + laneHalf) * 8 + vv] = (_Float16)e[col];
  }
  // col 64 = row feature (tile 4, n = 0); pad n = 1..15 of tile 4 with zeros
  dst[((4 * 32) + 0 + laneHalf) * 8 + vv] = (_Float16)(((float)r / (float)RD) * 2.f - 1.f);
#pragma unroll
  for (int nn = 1; nn < 16; ++nn)
    dst[((4 * 32) + nn + laneHalf) * 8 + vv] = (_Float16)0.f;

  if (c < 2) {
    const int tv = target[((b * CH + c) * RD + r) * SD + s];
    const float* te = emb + ((size_t)c * VD + tv) * ED;
    _Float16* td = tgtbuf + (size_t)(((c * RD + r) * SD + s) * 16 + b) * 64;
#pragma unroll
    for (int j = 0; j < 8; ++j) {
      f4v e0 = *(const f4v*)(te + j * 8);
      f4v e1 = *(const f4v*)(te + j * 8 + 4);
      v8h h8;
#pragma unroll
      for (int k = 0; k < 4; ++k) { h8[k] = (_Float16)e0[k]; h8[4 + k] = (_Float16)e1[k]; }
      *(v8h*)(td + j * 8) = h8;
    }
  }
}

// ---------------- wavefront GRU: one block per row, 1 compute wave ----------------

__global__ __launch_bounds__(128, 1) void rnn_kernel(
    const _Float16* __restrict__ exbuf,   // [RD][SD][5][32][8] (this channel, C-tile layout)
    const _Float16* __restrict__ pobuf,   // [RD][SD][16][32] prev-channel out, or null
    _Float16* __restrict__ outbuf,        // [RD][SD][16][32]
    const _Float16* __restrict__ zeroph,  // [16][32] zeros
    unsigned* __restrict__ flags,         // [RD][SD] pre-zeroed
    const float* __restrict__ Wih0,       // [96][65]
    const float* __restrict__ WihR,       // [2][96][32]
    const float* __restrict__ Whh,        // [3][96][32]
    const float* __restrict__ bih,        // [3][96]
    const float* __restrict__ bhh,        // [3][96]
    const float* __restrict__ h2eW,       // [65][32]
    const float* __restrict__ h2eb,       // [65]
    const float* __restrict__ adpW,       // [65][32] or null
    const float* __restrict__ adpb)       // [65] or null
{
  __shared__ _Float16 sWih0[NH * 64];     // k 0..63
  __shared__ float    sWih0c64[NH];       // k = 64 (row-feature column)
  __shared__ _Float16 sWihR[2 * NH * HD];
  __shared__ _Float16 sWhh[3 * NH * HD];
  __shared__ float    sBrz[3 * 64];       // bih+bhh for r,z gates
  __shared__ float    sBin[3 * HD];       // bih n-part
  __shared__ float    sBhn[3 * HD];       // bhh n-part
  __shared__ _Float16 sH2e[80 * HD];
  __shared__ float    sH2eb[80];
  __shared__ _Float16 sAdp[80 * HD];
  __shared__ float    sAdpb[80];
  __shared__ _Float16 sComb[16 * 64];
  __shared__ float    sCombC64[16];
  __shared__ _Float16 sHbuf[3 * 16 * HD];

  const int tid = threadIdx.x;
  const bool hasAdp = (adpW != nullptr);

  // ---- stage weights fp32 -> fp16 in LDS (all 128 threads) ----
  for (int i = tid; i < NH * 64; i += 128) { int nn = i >> 6, kk = i & 63; sWih0[i] = (_Float16)Wih0[nn * 65 + kk]; }
  for (int i = tid; i < NH; i += 128) sWih0c64[i] = Wih0[i * 65 + 64];
  for (int i = tid; i < 2 * NH * HD; i += 128) sWihR[i] = (_Float16)WihR[i];
  for (int i = tid; i < 3 * NH * HD; i += 128) sWhh[i] = (_Float16)Whh[i];
  for (int i = tid; i < 3 * 64; i += 128) { int l = i >> 6, j = i & 63; sBrz[i] = bih[l * NH + j] + bhh[l * NH + j]; }
  for (int i = tid; i < 3 * HD; i += 128) { int l = i / HD, j = i % HD; sBin[i] = bih[l * NH + 64 + j]; sBhn[i] = bhh[l * NH + 64 + j]; }
  for (int i = tid; i < 80 * HD; i += 128) { int nn = i >> 5, kk = i & 31; sH2e[i] = (nn < 65) ? (_Float16)h2eW[nn * HD + kk] : (_Float16)0.f; }
  for (int i = tid; i < 80; i += 128) sH2eb[i] = (i < 65) ? h2eb[i] : 0.f;
  for (int i = tid; i < 80 * HD; i += 128) { int nn = i >> 5, kk = i & 31; sAdp[i] = (hasAdp && nn < 65) ? (_Float16)adpW[nn * HD + kk] : (_Float16)0.f; }
  for (int i = tid; i < 80; i += 128) sAdpb[i] = (hasAdp && i < 65) ? adpb[i] : 0.f;
  __syncthreads();
  if (tid >= 32) return;   // wave 0 runs the sequential scan

  const int lane = tid;
  const int n    = lane & 15;
  const int mh   = (lane >> 4) * 8;
  const int r    = blockIdx.x;

  for (int i = lane; i < 3 * 16 * HD; i += 32) sHbuf[i] = (_Float16)0.f;
  v8f h[3][2];
#pragma unroll
  for (int l = 0; l < 3; ++l) { h[l][0] = splat8(0.f); h[l][1] = splat8(0.f); }

  volatile const unsigned* pflag =
      (r > 0) ? (volatile const unsigned*)(flags + (r - 1) * SD) : nullptr;

  for (int s = 0; s < SD; ++s) {
    // acquire previous-row output (wavefront dependency): tight spin, lowest latency
    if (r > 0) {
      while (pflag[s] == 0u) { }
      __threadfence();
    }
    const _Float16* ph = (r > 0) ? (outbuf + (size_t)((r - 1) * SD + s) * 512) : zeroph;
    const _Float16* ex = exbuf + (size_t)(r * SD + s) * 1280;

    v16h a_ph = load_a16(ph, HD, 0, lane);
    v16h a_po = a_ph;
    if (hasAdp) a_po = load_a16(pobuf + (size_t)(r * SD + s) * 512, HD, 0, lane);

    // comb = ex + ph@h2eW^T + h2eb (+ po@adpW^T + adpb); cols 0..64 live
#pragma unroll
    for (int t = 0; t < 5; ++t) {
      const float bias = sH2eb[t * 16 + n] + sAdpb[t * 16 + n];
      const v8h exv = *(const v8h*)(ex + (t * 32 + lane) * 8);   // C-tile layout: 1x b128
      v8f acc;
#pragma unroll
      for (int v = 0; v < 8; ++v) acc[v] = bias + (float)exv[v];
      acc = wmma16(a_ph, load_b16(sH2e, HD, t, 0, lane), acc);
      if (hasAdp) acc = wmma16(a_po, load_b16(sAdp, HD, t, 0, lane), acc);
      if (t < 4) {
        store_c16(sComb, 64, t, acc, lane);
      } else if (n == 0) {   // column 64 only (row feature)
#pragma unroll
        for (int v = 0; v < 8; ++v) sCombC64[mh + v] = acc[v];
      }
    }

    v16h a_c0 = load_a16(sComb, 64, 0, lane);
    v16h a_c1 = load_a16(sComb, 64, 32, lane);
    float c64a[8];
#pragma unroll
    for (int v = 0; v < 8; ++v) c64a[v] = sCombC64[mh + v];

#pragma unroll
    for (int l = 0; l < 3; ++l) {
      v16h a_h = load_a16(sHbuf + l * 16 * HD, HD, 0, lane);   // h_l from prev column
      v8f acc[6], ghn[2];
#pragma unroll
      for (int t = 0; t < 4; ++t) acc[t] = splat8(sBrz[l * 64 + t * 16 + n]);
#pragma unroll
      for (int i = 0; i < 2; ++i) {
        acc[4 + i] = splat8(sBin[l * HD + i * 16 + n]);
        ghn[i]     = splat8(sBhn[l * HD + i * 16 + n]);
      }
      // gi
      if (l == 0) {
#pragma unroll
        for (int t = 0; t < 6; ++t) {
          acc[t] = wmma16(a_c0, load_b16(sWih0, 64, t, 0, lane), acc[t]);
          acc[t] = wmma16(a_c1, load_b16(sWih0, 64, t, 32, lane), acc[t]);
          const float w64 = sWih0c64[t * 16 + n];        // rank-1 update for k=64
#pragma unroll
          for (int v = 0; v < 8; ++v) acc[t][v] += c64a[v] * w64;
        }
      } else {
        v16h a_in = load_a16(sHbuf + (l - 1) * 16 * HD, HD, 0, lane);
        const _Float16* wl = sWihR + (l - 1) * NH * HD;
#pragma unroll
        for (int t = 0; t < 6; ++t) acc[t] = wmma16(a_in, load_b16(wl, HD, t, 0, lane), acc[t]);
      }
      // gh: r,z accumulate directly; n-part kept separate for r*gh_n
      const _Float16* whl = sWhh + l * NH * HD;
#pragma unroll
      for (int t = 0; t < 4; ++t) acc[t] = wmma16(a_h, load_b16(whl, HD, t, 0, lane), acc[t]);
#pragma unroll
      for (int i = 0; i < 2; ++i) ghn[i] = wmma16(a_h, load_b16(whl, HD, 4 + i, 0, lane), ghn[i]);

      // gates: lane-aligned across tiles (pure VALU, no shuffles)
#pragma unroll
      for (int i = 0; i < 2; ++i) {
        v8f rr = sigm8(acc[i]);
        v8f zz = sigm8(acc[2 + i]);
        v8f nn_;
#pragma unroll
        for (int v = 0; v < 8; ++v) nn_[v] = tanh_fast(acc[4 + i][v] + rr[v] * ghn[i][v]);
        v8f hn;
#pragma unroll
        for (int v = 0; v < 8; ++v) hn[v] = (1.f - zz[v]) * nn_[v] + zz[v] * h[l][i][v];
        h[l][i] = hn;
        store_c16(sHbuf + l * 16 * HD, HD, i, hn, lane);   // A-layout staging for next use
      }
    }

    // top-layer output -> global (f16, m-major) + release flag for row r+1
    _Float16* o = outbuf + (size_t)(r * SD + s) * 512;
#pragma unroll
    for (int i = 0; i < 2; ++i)
#pragma unroll
      for (int v = 0; v < 8; ++v) o[(mh + v) * HD + i * 16 + n] = (_Float16)h[2][i][v];
    __threadfence();
    if (lane == 0) ((volatile unsigned*)flags)[r * SD + s] = 1u;
  }
}

// ---------------- logit heads: 16x{32,96,160} @ {D}x258 per pixel ----------------

__global__ __launch_bounds__(256, 1) void logits_kernel(
    const _Float16* __restrict__ outsbuf,  // [3][RD][SD][16][32]
    const _Float16* __restrict__ tgtbuf,   // [2][RD][SD][16][64]
    const float* __restrict__ redW,  const float* __restrict__ redb,
    const float* __restrict__ greenW, const float* __restrict__ greenb,
    const float* __restrict__ blueW, const float* __restrict__ blueb,
    float* __restrict__ out)               // [BD][3][RD][SD][VD]
{
  __shared__ _Float16 slab[272 * 32];      // K-slab of W, rows padded 258->272
  const int tid  = threadIdx.x;
  const int lane = tid & 31, w = tid >> 5;
  const int n = lane & 15, mh = (lane >> 4) * 8;

  int blk = blockIdx.x;
  const int c = blk / (RD * 8); blk %= RD * 8;
  const int r = blk >> 3;
  const int s = (blk & 7) * 8 + w;

  const float* W  = (c == 0) ? redW : (c == 1) ? greenW : blueW;
  const float* bb = (c == 0) ? redb : (c == 1) ? greenb : blueb;
  const int D   = (c == 0) ? HD : (c == 1) ? (HD + ED) : (HD + 2 * ED);
  const int nkt = D >> 5;

  v8f acc[17];
#pragma unroll
  for (int t = 0; t < 17; ++t) {
    const int vc = t * 16 + n;
    acc[t] = splat8(vc < VD ? bb[vc] : 0.f);
  }

  for (int kt = 0; kt < nkt; ++kt) {
    __syncthreads();
    // stage W K-slab: float4 loads -> 64-bit f16 ds stores
    for (int i = tid; i < 272 * 8; i += 256) {
      const int row = i >> 3, cg = (i & 7) * 4;
      v4h h4;
      if (row < VD) {
        f4v w4 = *(const f4v*)(W + (size_t)row * D + kt * 32 + cg);
#pragma unroll
        for (int k = 0; k < 4; ++k) h4[k] = (_Float16)w4[k];
      } else {
#pragma unroll
        for (int k = 0; k < 4; ++k) h4[k] = (_Float16)0.f;
      }
      *(v4h*)(slab + row * 32 + cg) = h4;
    }
    __syncthreads();

    const _Float16* ap; int ld, k0;
    if (kt == 0)      { ap = outsbuf + (size_t)((c * RD + r) * SD + s) * 512;  ld = HD; k0 = 0; }
    else if (kt <= 2) { ap = tgtbuf  + (size_t)((0 * RD + r) * SD + s) * 1024; ld = 64; k0 = (kt - 1) * 32; }
    else              { ap = tgtbuf  + (size_t)((1 * RD + r) * SD + s) * 1024; ld = 64; k0 = (kt - 3) * 32; }
    v16h a = load_a16(ap, ld, k0, lane);
#pragma unroll
    for (int t = 0; t < 17; ++t) acc[t] = wmma16(a, load_b16(slab, 32, t, 0, lane), acc[t]);
  }

#pragma unroll
  for (int t = 0; t < 17; ++t) {
    const int vc = t * 16 + n;
    if (vc < VD) {
#pragma unroll
      for (int v = 0; v < 8; ++v) {
        const int b = mh + v;
        out[(size_t)(((b * 3 + c) * RD + r) * SD + s) * VD + vc] = acc[t][v];
      }
    }
  }
}

// ---------------- host launch ----------------

extern "C" void kernel_launch(void* const* d_in, const int* in_sizes, int n_in,
                              void* d_out, int out_size, void* d_ws, size_t ws_size,
                              hipStream_t stream) {
  (void)in_sizes; (void)n_in; (void)out_size; (void)ws_size;
  const int*   x      = (const int*)d_in[0];
  const int*   target = (const int*)d_in[1];
  const float* emb    = (const float*)d_in[2];
  const float* Wih0   = (const float*)d_in[3];
  const float* WihR   = (const float*)d_in[4];
  const float* Whh    = (const float*)d_in[5];
  const float* bih    = (const float*)d_in[6];
  const float* bhh    = (const float*)d_in[7];
  const float* h2eW   = (const float*)d_in[8];
  const float* h2eb   = (const float*)d_in[9];
  const float* adpW   = (const float*)d_in[10];
  const float* adpb   = (const float*)d_in[11];
  const float* redW   = (const float*)d_in[12];
  const float* redb   = (const float*)d_in[13];
  const float* greenW = (const float*)d_in[14];
  const float* greenb = (const float*)d_in[15];
  const float* blueW  = (const float*)d_in[16];
  const float* blueb  = (const float*)d_in[17];
  float* out = (float*)d_out;

  char* p = (char*)d_ws;
  _Float16* exbuf   = (_Float16*)p; p += (size_t)CH * RD * SD * 1280 * sizeof(_Float16);
  _Float16* outsbuf = (_Float16*)p; p += (size_t)CH * RD * SD * 512 * sizeof(_Float16);
  _Float16* tgtbuf  = (_Float16*)p; p += (size_t)2 * RD * SD * 1024 * sizeof(_Float16);
  _Float16* zeroph  = (_Float16*)p; p += (size_t)512 * sizeof(_Float16);
  unsigned* flags   = (unsigned*)p; p += (size_t)CH * RD * SD * sizeof(unsigned);

  init_kernel<<<48, 256, 0, stream>>>(flags, zeroph);
  embed_kernel<<<(CH * RD * SD * BD + 255) / 256, 256, 0, stream>>>(x, target, emb, exbuf, tgtbuf);

  for (int c = 0; c < CH; ++c) {
    rnn_kernel<<<RD, 128, 0, stream>>>(
        exbuf + (size_t)c * RD * SD * 1280,
        (c > 0) ? outsbuf + (size_t)(c - 1) * RD * SD * 512 : nullptr,
        outsbuf + (size_t)c * RD * SD * 512,
        zeroph,
        flags + c * RD * SD,
        Wih0 + (size_t)c * 96 * 65,
        WihR + (size_t)c * 2 * 96 * 32,
        Whh  + (size_t)c * 3 * 96 * 32,
        bih  + (size_t)c * 3 * 96,
        bhh  + (size_t)c * 3 * 96,
        h2eW + (size_t)c * 65 * 32,
        h2eb + (size_t)c * 65,
        (c > 0) ? adpW + (size_t)(c - 1) * 65 * 32 : nullptr,
        (c > 0) ? adpb + (size_t)(c - 1) * 65 : nullptr);
  }

  logits_kernel<<<CH * RD * 8, 256, 0, stream>>>(outsbuf, tgtbuf, redW, redb,
                                                 greenW, greenb, blueW, blueb, out);
}
